// DiscreteMessagePassing_29703993819531
// MI455X (gfx1250) — matrix operand order repaired and checked by reading, not compiled, not run on hardware
//
#include <hip/hip_runtime.h>
#include <hip/hip_bf16.h>
#include <math.h>

#define N_NODES 50000
#define N_EDGES 800000
#define IN_F 64
#define OUT_F 64
#define HID 128
#define MSG 64
#define INV_TAU 10.0f
#define G_EPS 1e-10f

typedef __attribute__((ext_vector_type(16))) __bf16 v16bf;
typedef __attribute__((ext_vector_type(8)))  float  v8f;
typedef __attribute__((ext_vector_type(2)))  float  f2;

// ---------------- WMMA fragment helpers (contiguous k order: k = 16*half + e) ----------

// A-matrix 16x32 bf16 from row-major LDS tile [16][ld]
__device__ __forceinline__ v16bf load_A_frag(const __bf16* t, int ld) {
  int lane = threadIdx.x & 31;
  int half = lane >> 4, r = lane & 15;
  v16bf a;
#pragma unroll
  for (int i = 0; i < 8; ++i) {
    int k = half * 16 + 2 * i;
    a[2 * i]     = t[r * ld + k];
    a[2 * i + 1] = t[r * ld + k + 1];
  }
  return a;
}

// B-matrix fragment from pre-swizzled weights: tile (kc,nc) stored as
// 32 lanes x 16 contiguous bf16 (32B per lane -> two global_load_b128)
__device__ __forceinline__ v16bf load_B_packed(const __bf16* packed, int kc,
                                               int nc, int KC) {
  int lane = threadIdx.x & 31;
  return *(const v16bf*)(packed +
                         ((size_t)(nc * KC + kc) * 512 + (size_t)lane * 16));
}

// Native CDNA5 float-max atomic (segment_max), device scope
__device__ __forceinline__ void global_atomic_fmax(float* addr, float val) {
  asm volatile("global_atomic_max_num_f32 %0, %1, off scope:SCOPE_DEV"
               :: "v"(addr), "v"(val) : "memory");
}

// gfx1250 async global->LDS copy (ASYNCcnt tracked, no VGPR round-trip).
// LDS byte offset = low 32 bits of the generic shared-pointer address
// (flat-LDS aperture mapping, ISA section 10.2).
__device__ __forceinline__ void async_copy_b128(void* lds_ptr, const void* g) {
  unsigned lds_off = (unsigned)(unsigned long long)lds_ptr;
  asm volatile("global_load_async_to_lds_b128 %0, %1, off"
               :: "v"(lds_off), "v"(g) : "memory");
}
__device__ __forceinline__ void wait_async0() {
  asm volatile("s_wait_asynccnt 0x0" ::: "memory");
}

// ---------------- prep kernels ----------------------------------------------

__global__ void f32_to_bf16_kernel(const float* __restrict__ src,
                                   __bf16* __restrict__ dst, int n) {
  for (int i = blockIdx.x * blockDim.x + threadIdx.x; i < n;
       i += gridDim.x * blockDim.x)
    dst[i] = (__bf16)src[i];
}

// Swizzle a weight matrix into per-tile WMMA B-fragment layout.
// dst element index = ((nc*KC + kc)*32 + lane)*16 + e,
// with lane = {half,n}, k = half*16 + e  (B VGPR pairs hold consecutive K).
// transposed=0: src is [K][N] row-major;  transposed=1: src is [N][K] (torch).
__global__ void pack_B_kernel(const float* __restrict__ src,
                              __bf16* __restrict__ dst, int K, int N,
                              int transposed) {
  int KC = K >> 5, NC = N >> 4;
  int total = KC * NC * 512;
  for (int idx = blockIdx.x * blockDim.x + threadIdx.x; idx < total;
       idx += gridDim.x * blockDim.x) {
    int t = idx >> 9;
    int l = (idx >> 4) & 31;
    int e = idx & 15;
    int nc = t / KC, kc = t - nc * KC;
    int half = l >> 4, n = l & 15;
    int gk = kc * 32 + half * 16 + e;
    int gn = nc * 16 + n;
    float v = transposed ? src[(size_t)gn * K + gk] : src[(size_t)gk * N + gn];
    dst[idx] = (__bf16)v;
  }
}

// ---------------- kernel A: per-node encoder logits --------------------------
// L[N,64] = relu(x @ W1 + b1) @ W2 + b2 ; 16 rows / block, 8 waves

__global__ __launch_bounds__(256) void encoder_kernel(
    const __bf16* __restrict__ xbf,
    const __bf16* __restrict__ W1p, const float* __restrict__ b1,
    const __bf16* __restrict__ W2p, const float* __restrict__ b2,
    float* __restrict__ L) {
  __shared__ __bf16 xT[16 * 72];    // padded ld=72 (144B rows, 16B aligned)
  __shared__ __bf16 hT[16 * 136];   // padded ld=136
  const int tid = threadIdx.x;
  const int wave = tid >> 5, lane = tid & 31;
  const int row0 = blockIdx.x * 16;

  if (tid < 128) {                  // 16 rows x 8 x 16B segments, async
    int r = tid >> 3, seg = tid & 7;
    async_copy_b128(&xT[r * 72 + seg * 8],
                    xbf + (size_t)(row0 + r) * 64 + seg * 8);
  }
  wait_async0();
  __syncthreads();

  // H = relu(X @ W1 + b1): [16,128]; wave w -> cols [16w,16w+16), K=64
  {
    v8f acc = {};
#pragma unroll
    for (int kc = 0; kc < 2; ++kc) {
      v16bf a = load_A_frag(xT + kc * 32, 72);
      v16bf b = load_B_packed(W1p, kc, wave, 2);
      acc = __builtin_amdgcn_wmma_f32_16x16x32_bf16(false, a, false, b,
                                                    (short)0, acc, false, false);
    }
    int half = lane >> 4, c = lane & 15;
    int col = wave * 16 + c;
    float bias = b1[col];
#pragma unroll
    for (int v = 0; v < 8; ++v)
      hT[(v + 8 * half) * 136 + col] = (__bf16)fmaxf(acc[v] + bias, 0.0f);
  }
  __syncthreads();

  // L = H @ W2 + b2: [16,64]; waves 0..3, K=128
  if (wave < 4) {
    v8f acc = {};
#pragma unroll
    for (int kc = 0; kc < 4; ++kc) {
      v16bf a = load_A_frag(hT + kc * 32, 136);
      v16bf b = load_B_packed(W2p, kc, wave, 4);
      acc = __builtin_amdgcn_wmma_f32_16x16x32_bf16(false, a, false, b,
                                                    (short)0, acc, false, false);
    }
    int half = lane >> 4, c = lane & 15;
    int col = wave * 16 + c;
    float bias = b2[col];
#pragma unroll
    for (int v = 0; v < 8; ++v)
      L[(size_t)(row0 + v + 8 * half) * 64 + col] = acc[v] + bias;
  }
}

// ---------------- kernel B: per-edge gumbel-softmax + scatter max ------------
// one wave32 per edge, 2 consecutive msg elems per lane

__global__ __launch_bounds__(256) void edge_kernel(
    const float* __restrict__ L, const int* __restrict__ src,
    const int* __restrict__ dst, const float* __restrict__ u,
    float* __restrict__ y) {
  const int wave = threadIdx.x >> 5, lane = threadIdx.x & 31;
  const int e = blockIdx.x * 8 + wave;
  if (e >= N_EDGES) return;
  const int s = src[e], d = dst[e];
  f2 l2 = *(const f2*)(L + (size_t)s * MSG + 2 * lane);
  // u is a 205MB single-use stream: non-temporal, keep L/y resident in L2
  f2 u2 = __builtin_nontemporal_load((const f2*)(u + (size_t)e * MSG + 2 * lane));
  float g0 = -__logf(-__logf(u2.x + G_EPS) + G_EPS);
  float g1 = -__logf(-__logf(u2.y + G_EPS) + G_EPS);
  float t0 = (l2.x + g0) * INV_TAU;
  float t1 = (l2.y + g1) * INV_TAU;
  float mx = fmaxf(t0, t1);
#pragma unroll
  for (int off = 16; off > 0; off >>= 1) mx = fmaxf(mx, __shfl_xor(mx, off, 32));
  float e0 = __expf(t0 - mx), e1 = __expf(t1 - mx);
  float sm = e0 + e1;
#pragma unroll
  for (int off = 16; off > 0; off >>= 1) sm += __shfl_xor(sm, off, 32);
  float inv = 1.0f / sm;
  float* yrow = y + (size_t)d * MSG + 2 * lane;
  global_atomic_fmax(yrow,     e0 * inv);
  global_atomic_fmax(yrow + 1, e1 * inv);
}

// ---------------- kernel C: fused decode + GRU update ------------------------
// 16 rows / block, 8 waves

__global__ __launch_bounds__(256) void update_kernel(
    const float* __restrict__ y,
    const __bf16* __restrict__ xbf, const __bf16* __restrict__ zbf,
    const float* __restrict__ z,
    const __bf16* __restrict__ Wdecp, const float* __restrict__ bdec,
    const __bf16* __restrict__ Wihp,  const float* __restrict__ bih,
    const __bf16* __restrict__ Whhp,  const float* __restrict__ bhh,
    float* __restrict__ out) {
  __shared__ __bf16 yT[16 * 72];
  __shared__ __bf16 xT[16 * 72];
  __shared__ __bf16 zT[16 * 72];
  __shared__ __bf16 decT[16 * 136];
  __shared__ float  giT[16 * 192];
  __shared__ float  ghT[16 * 192];
  const int tid = threadIdx.x;
  const int wave = tid >> 5, lane = tid & 31;
  const int row0 = blockIdx.x * 16;

  // x/z tiles: pure byte copies -> async LDS path
  {
    int t = tid & 127;
    int r = t >> 3, seg = t & 7;
    if (tid < 128)
      async_copy_b128(&xT[r * 72 + seg * 8],
                      xbf + (size_t)(row0 + r) * 64 + seg * 8);
    else
      async_copy_b128(&zT[r * 72 + seg * 8],
                      zbf + (size_t)(row0 + r) * 64 + seg * 8);
  }
  // y tile needs f32->bf16 conversion: VALU path
  for (int idx = tid; idx < 16 * 64; idx += 256) {
    int r = idx >> 6, c = idx & 63;
    yT[r * 72 + c] = (__bf16)y[(size_t)(row0 + r) * 64 + c];
  }
  wait_async0();
  __syncthreads();

  // dec = relu(y @ Wdec + bdec): [16,128]; wave w -> col tile w, K=64
  {
    v8f acc = {};
#pragma unroll
    for (int kc = 0; kc < 2; ++kc) {
      v16bf a = load_A_frag(yT + kc * 32, 72);
      v16bf b = load_B_packed(Wdecp, kc, wave, 2);
      acc = __builtin_amdgcn_wmma_f32_16x16x32_bf16(false, a, false, b,
                                                    (short)0, acc, false, false);
    }
    int half = lane >> 4, c = lane & 15;
    int col = wave * 16 + c;
    float bias = bdec[col];
#pragma unroll
    for (int v = 0; v < 8; ++v)
      decT[(v + 8 * half) * 136 + col] = (__bf16)fmaxf(acc[v] + bias, 0.0f);
  }
  __syncthreads();

  // jobs 0..11:  gi = [x|dec] @ Wih^T + bih  (K=192, 6 WMMAs)
  // jobs 12..23: gh = z @ Whh^T + bhh        (K=64, 2 WMMAs)
  for (int j = wave; j < 24; j += 8) {
    v8f acc = {};
    int half = lane >> 4, c = lane & 15;
    if (j < 12) {
#pragma unroll
      for (int kc = 0; kc < 6; ++kc) {
        v16bf a = (kc < 2) ? load_A_frag(xT + kc * 32, 72)
                           : load_A_frag(decT + (kc - 2) * 32, 136);
        v16bf b = load_B_packed(Wihp, kc, j, 6);
        acc = __builtin_amdgcn_wmma_f32_16x16x32_bf16(false, a, false, b,
                                                      (short)0, acc, false, false);
      }
      int col = j * 16 + c;
      float bias = bih[col];
#pragma unroll
      for (int v = 0; v < 8; ++v) giT[(v + 8 * half) * 192 + col] = acc[v] + bias;
    } else {
      int jj = j - 12;
#pragma unroll
      for (int kc = 0; kc < 2; ++kc) {
        v16bf a = load_A_frag(zT + kc * 32, 72);
        v16bf b = load_B_packed(Whhp, kc, jj, 2);
        acc = __builtin_amdgcn_wmma_f32_16x16x32_bf16(false, a, false, b,
                                                      (short)0, acc, false, false);
      }
      int col = jj * 16 + c;
      float bias = bhh[col];
#pragma unroll
      for (int v = 0; v < 8; ++v) ghT[(v + 8 * half) * 192 + col] = acc[v] + bias;
    }
  }
  __syncthreads();

  // GRU gates, gate order r,z,n; write both tuple halves of the output
  for (int idx = tid; idx < 16 * 64; idx += 256) {
    int r = idx >> 6, c = idx & 63;
    float ir = giT[r * 192 + c], iz = giT[r * 192 + 64 + c],
          in_ = giT[r * 192 + 128 + c];
    float hr = ghT[r * 192 + c], hz = ghT[r * 192 + 64 + c],
          hn = ghT[r * 192 + 128 + c];
    float rg = 1.0f / (1.0f + __expf(-(ir + hr)));
    float zg = 1.0f / (1.0f + __expf(-(iz + hz)));
    float ng = tanhf(in_ + rg * hn);
    size_t gidx = (size_t)(row0 + r) * 64 + c;
    float hprev = z[gidx];
    float h = (1.0f - zg) * ng + zg * hprev;
    out[gidx] = h;
    out[(size_t)N_NODES * 64 + gidx] = h;
  }
}

// ---------------- launch -----------------------------------------------------

extern "C" void kernel_launch(void* const* d_in, const int* in_sizes, int n_in,
                              void* d_out, int out_size, void* d_ws, size_t ws_size,
                              hipStream_t stream) {
  (void)in_sizes; (void)n_in; (void)out_size; (void)ws_size;
  const float* x      = (const float*)d_in[0];
  const float* z      = (const float*)d_in[1];
  const int*   src    = (const int*)d_in[2];
  const int*   dst    = (const int*)d_in[3];
  const float* u      = (const float*)d_in[4];
  const float* W_enc1 = (const float*)d_in[5];
  const float* b_enc1 = (const float*)d_in[6];
  const float* W_enc2 = (const float*)d_in[7];
  const float* b_enc2 = (const float*)d_in[8];
  const float* W_dec  = (const float*)d_in[9];
  const float* b_dec  = (const float*)d_in[10];
  const float* W_ih   = (const float*)d_in[11];
  const float* b_ih   = (const float*)d_in[12];
  const float* W_hh   = (const float*)d_in[13];
  const float* b_hh   = (const float*)d_in[14];

  char* ws = (char*)d_ws;
  size_t off = 0;
  auto alloc = [&](size_t bytes) {
    void* p = ws + off;
    off += (bytes + 255) & ~(size_t)255;
    return p;
  };
  __bf16* xbf   = (__bf16*)alloc((size_t)N_NODES * 64 * sizeof(__bf16));
  __bf16* zbf   = (__bf16*)alloc((size_t)N_NODES * 64 * sizeof(__bf16));
  float*  Lbuf  = (float*)alloc((size_t)N_NODES * 64 * sizeof(float));
  float*  ybuf  = (float*)alloc((size_t)N_NODES * 64 * sizeof(float));
  __bf16* W1p   = (__bf16*)alloc((size_t)IN_F * HID * sizeof(__bf16));
  __bf16* W2p   = (__bf16*)alloc((size_t)HID * MSG * sizeof(__bf16));
  __bf16* Wdecp = (__bf16*)alloc((size_t)MSG * HID * sizeof(__bf16));
  __bf16* Wihp  = (__bf16*)alloc((size_t)3 * OUT_F * (IN_F + HID) * sizeof(__bf16));
  __bf16* Whhp  = (__bf16*)alloc((size_t)3 * OUT_F * OUT_F * sizeof(__bf16));

  auto conv = [&](const float* s, __bf16* dct, int n) {
    int grid = (n + 1023) / 1024;
    if (grid > 4096) grid = 4096;
    f32_to_bf16_kernel<<<grid, 256, 0, stream>>>(s, dct, n);
  };
  conv(x, xbf, N_NODES * 64);
  conv(z, zbf, N_NODES * 64);

  auto pack = [&](const float* s, __bf16* dct, int K, int N, int tr) {
    int n = (K >> 5) * (N >> 4) * 512;
    pack_B_kernel<<<(n + 255) / 256, 256, 0, stream>>>(s, dct, K, N, tr);
  };
  pack(W_enc1, W1p,   IN_F, HID, 0);   // [K=64][N=128] row-major
  pack(W_enc2, W2p,   HID,  MSG, 0);   // [K=128][N=64] row-major
  pack(W_dec,  Wdecp, MSG,  HID, 0);   // [K=64][N=128] row-major
  pack(W_ih,   Wihp,  IN_F + HID, 3 * OUT_F, 1);  // torch [N=192][K=192]
  pack(W_hh,   Whhp,  OUT_F,      3 * OUT_F, 1);  // torch [N=192][K=64]

  // segment_max identity: softmax outputs > 0, so y=0 matches the
  // "zero in-degree -> 0" semantics of the reference
  hipMemsetAsync(ybuf, 0, (size_t)N_NODES * 64 * sizeof(float), stream);

  encoder_kernel<<<N_NODES / 16, 256, 0, stream>>>(xbf, W1p, b_enc1, W2p,
                                                   b_enc2, Lbuf);
  edge_kernel<<<N_EDGES / 8, 256, 0, stream>>>(Lbuf, src, dst, u, ybuf);
  update_kernel<<<N_NODES / 16, 256, 0, stream>>>(ybuf, xbf, zbf, z, Wdecp,
                                                  b_dec, Wihp, b_ih, Whhp,
                                                  b_hh, (float*)d_out);
}
